// KnowledgeEmbedding_62646392979850
// MI455X (gfx1250) — matrix operand order repaired and checked by reading, not compiled, not run on hardware
//
#include <hip/hip_runtime.h>
#include <math.h>

#define NF   20000
#define NH   40000
#define DIM  128
#define ADIM 64
#define BATCH 256
#define NE   640000
#define KCHUNKS 625   // NF / 32
#define KSPLIT  125   // 625 = 125 * 5
#define KC_PER  5
#define SCH  16       // connections per scatter block

typedef __attribute__((ext_vector_type(16))) _Float16 v16h;
typedef __attribute__((ext_vector_type(16))) __bf16   v16bf;
typedef __attribute__((ext_vector_type(8)))  float    v8f;
typedef __attribute__((ext_vector_type(4)))  int      v4i;

typedef __attribute__((address_space(1))) v4i* gv4i_ptr;   // global v4i*
typedef __attribute__((address_space(3))) v4i* lv4i_ptr;   // LDS v4i*

__device__ __forceinline__ unsigned short f2bf(float f) {
    unsigned int u = __float_as_uint(f);
    u += 0x7FFFu + ((u >> 16) & 1u);       // round-to-nearest-even
    return (unsigned short)(u >> 16);
}

// ---------------------------------------------------------------------------
// Async global -> LDS copy of 16 bytes per lane (CDNA5 GLOBAL_LOAD_ASYNC_TO_LDS_B128,
// tracked by ASYNCcnt). Builtin if the toolchain has it, else inline asm per
// CDNA5 ISA.
#if defined(__has_builtin) && __has_builtin(__builtin_amdgcn_global_load_async_to_lds_b128)
#define ATHENA_ASYNC_BUILTIN 1
#else
#define ATHENA_ASYNC_BUILTIN 0
#endif

__device__ __forceinline__ void async_copy16(const void* gsrc, void* ldst) {
#if ATHENA_ASYNC_BUILTIN
    __builtin_amdgcn_global_load_async_to_lds_b128(
        (gv4i_ptr)gsrc, (lv4i_ptr)ldst, 0, 0);
#else
    unsigned lds32 = (unsigned)(unsigned long long)ldst;  // LDS aperture: low 32 bits = offset
    asm volatile("global_load_async_to_lds_b128 %0, %1, off"
                 :: "v"(lds32), "v"((unsigned long long)gsrc)
                 : "memory");
#endif
}

__device__ __forceinline__ void wait_async0() {
#if defined(__has_builtin) && __has_builtin(__builtin_amdgcn_s_wait_asynccnt)
    __builtin_amdgcn_s_wait_asynccnt(0);
#else
    asm volatile("s_wait_asynccnt 0x0" ::: "memory");
#endif
}

__device__ __forceinline__ void wait_async1() {
#if defined(__has_builtin) && __has_builtin(__builtin_amdgcn_s_wait_asynccnt)
    __builtin_amdgcn_s_wait_asynccnt(1);
#else
    asm volatile("s_wait_asynccnt 0x1" ::: "memory");
#endif
}

// ---------------------------------------------------------------------------
// zero a float region
__global__ void zero_kernel(float* __restrict__ p, long long n) {
    long long i = (long long)blockIdx.x * blockDim.x + threadIdx.x;
    if (i < n) p[i] = 0.0f;
}

// ---------------------------------------------------------------------------
// Attention scores: one wave handles 16 connections as a 16x256x64 GEMM strip
// using v_wmma_f32_16x16x32_f16. Epilogue: tanh, dot with u, exp, corrective,
// atomic add into segment denominators.
__global__ void __launch_bounds__(256)
attn_scores_kernel(const float* __restrict__ feat,
                   const float* __restrict__ hid,
                   const float* __restrict__ Wk,     // [256 x 64] row-major
                   const float* __restrict__ bias,   // [64]
                   const float* __restrict__ u,      // [64]
                   const float* __restrict__ corr,   // [E]
                   const int*   __restrict__ part,   // [E]
                   const int*   __restrict__ conn,   // [E]
                   float* __restrict__ scores,       // [E]
                   float* __restrict__ denom)        // [F]
{
    // Pre-swizzled f16 B-fragments of W: [kchunk 8][ntile 4][lane 32][half 16]
    __shared__ __align__(32) _Float16 bfragW[8][4][32][16];

    const int tid = threadIdx.x;
    for (int i = tid; i < 8 * 4 * 32 * 16; i += 256) {
        int h    = i & 15;
        int lane = (i >> 4) & 31;
        int t    = (i >> 9) & 3;
        int c    = i >> 11;
        int k = 32 * c + (h & 7) + ((lane >> 4) << 3) + ((h >> 3) << 4);   // B layout: lanes 0-15 K {0..7,16..23}, 16-31 K {8..15,24..31}
        int n = 16 * t + (lane & 15);
        bfragW[c][t][lane][h] = (_Float16)Wk[k * ADIM + n];
    }
    __syncthreads();

    const int wave = tid >> 5;
    const int lane = tid & 31;
    const int m    = lane & 15;
    const int hi   = lane >> 4;
    const long long e0 = (long long)(blockIdx.x * 8 + wave) * 16;

    const int p  = part[e0 + m];
    const int ci = conn[e0 + m];
    const float* rowF = feat + (long long)p * DIM;
    const float* rowC = (ci < NF) ? (feat + (long long)ci * DIM)
                                  : (hid + (long long)(ci - NF) * DIM);

    const v8f vz = {0, 0, 0, 0, 0, 0, 0, 0};
    v8f acc[4] = {vz, vz, vz, vz};

    const int off0 = hi ? 8 : 0;   // A layout: lanes 0-15 take K {0..7,16..23}, 16-31 take {8..15,24..31}
#pragma unroll
    for (int c = 0; c < 8; ++c) {
        const float* base = (c < 4) ? (rowF + c * 32) : (rowC + (c - 4) * 32);
        union { v16h v; _Float16 h[16]; } a;
#pragma unroll
        for (int j = 0; j < 8; ++j) a.h[j]     = (_Float16)base[off0 + j];
#pragma unroll
        for (int j = 0; j < 8; ++j) a.h[8 + j] = (_Float16)base[off0 + 16 + j];
#pragma unroll
        for (int t = 0; t < 4; ++t) {
            v16h b = *reinterpret_cast<const v16h*>(&bfragW[c][t][lane][0]);
            acc[t] = __builtin_amdgcn_wmma_f32_16x16x32_f16(
                false, a.v, false, b, (short)0, acc[t], false, false);
        }
    }

    // Epilogue: h = tanh(acc + bias), partial dot with u per lane-column.
    float partial[8];
#pragma unroll
    for (int r = 0; r < 8; ++r) partial[r] = 0.0f;
#pragma unroll
    for (int t = 0; t < 4; ++t) {
        int n = 16 * t + m;
        float bt = bias[n];
        float ut = u[n];
#pragma unroll
        for (int r = 0; r < 8; ++r)
            partial[r] += tanhf(acc[t][r] + bt) * ut;
    }
    // reduce across the 16 lanes that share a row group
#pragma unroll
    for (int mask = 1; mask < 16; mask <<= 1) {
#pragma unroll
        for (int r = 0; r < 8; ++r)
            partial[r] += __shfl_xor(partial[r], mask, 32);
    }
    // lanes 0-7 emit rows 0-7, lanes 16-23 emit rows 8-15
    if ((lane & 15) < 8) {
        int r  = lane & 7;
        int m2 = r + 8 * hi;
        long long e = e0 + m2;
        float s = 0.0f;
#pragma unroll
        for (int rr = 0; rr < 8; ++rr) if (rr == r) s = partial[rr];
        s = expf(s) * corr[e];
        scores[e] = s;
        atomicAdd(&denom[part[e]], s);
    }
}

// ---------------------------------------------------------------------------
// weights = scores/denom[p]; context[p] += w * conn_emb. Sorted partition ->
// run-length compress the atomics.
__global__ void __launch_bounds__(128)
scatter_context_kernel(const float* __restrict__ scores,
                       const float* __restrict__ denom,
                       const int*   __restrict__ part,
                       const int*   __restrict__ conn,
                       const float* __restrict__ feat,
                       const float* __restrict__ hid,
                       float* __restrict__ ctx)     // [F x 128]
{
    const int d = threadIdx.x;                      // 0..127
    const long long base = (long long)blockIdx.x * SCH;
    float accv = 0.0f;
    int curp = -1;
    for (int i = 0; i < SCH; ++i) {
        long long e = base + i;
        int p  = part[e];
        int ci = conn[e];
        float w = scores[e] / denom[p];
        const float* row = (ci < NF) ? (feat + (long long)ci * DIM)
                                     : (hid + (long long)(ci - NF) * DIM);
        float v = w * row[d];
        if (p != curp) {
            if (curp >= 0) atomicAdd(&ctx[(long long)curp * DIM + d], accv);
            curp = p;
            accv = v;
        } else {
            accv += v;
        }
    }
    if (curp >= 0) atomicAdd(&ctx[(long long)curp * DIM + d], accv);
}

// ---------------------------------------------------------------------------
// pack values f32 -> bf16, row-major [B x F]
__global__ void pack_values_kernel(const float* __restrict__ vals,
                                   unsigned short* __restrict__ out)
{
    long long i = ((long long)blockIdx.x * blockDim.x + threadIdx.x) * 4;
    float4 f = *reinterpret_cast<const float4*>(vals + i);
    ushort4 o;
    o.x = f2bf(f.x); o.y = f2bf(f.y); o.z = f2bf(f.z); o.w = f2bf(f.w);
    *reinterpret_cast<ushort4*>(out + i) = o;
}

// ---------------------------------------------------------------------------
// swizzle context f32 [F x 128] -> bf16 WMMA B-fragment layout
// ctxB[((kc*8 + t)*32 + lane)*16 + h] = bf16(ctx[(32*kc + (h&7) + 8*(lane>>4) + 16*(h>>3)) * 128 + 16*t + (lane&15)])
__global__ void pack_ctxB_kernel(const float* __restrict__ ctx,
                                 unsigned short* __restrict__ ctxB)
{
    int i = blockIdx.x * blockDim.x + threadIdx.x;   // [0, F*D)
    int h    = i & 15;
    int lane = (i >> 4) & 31;
    int t    = (i >> 9) & 7;
    int kc   = i >> 12;
    int k = 32 * kc + (h & 7) + ((lane >> 4) << 3) + ((h >> 3) << 4);
    int n = 16 * t + (lane & 15);
    ctxB[i] = f2bf(ctx[(long long)k * DIM + n]);
}

// ---------------------------------------------------------------------------
// out[256x128] = values[256x20000] @ context[20000x128] in bf16 WMMA.
// 125 K-slices; one block (16 waves = all 16 M-tiles) per slice; B fragments
// double-buffered through LDS with GLOBAL_LOAD_ASYNC_TO_LDS_B128 so each
// 8 KB K-chunk is fetched once per block instead of once per wave.
__global__ void __launch_bounds__(512)
final_gemm_kernel(const unsigned short* __restrict__ valsB,  // bf16 [B x F]
                  const unsigned short* __restrict__ ctxB,   // bf16 B-fragments
                  float* __restrict__ out)                   // [B x 128]
{
    __shared__ __align__(32) unsigned short Bbuf[2][4096];   // 2 x 8 KB

    const int tid  = threadIdx.x;       // 0..511
    const int wave = tid >> 5;          // 0..15 -> M tile
    const int lane = tid & 31;
    const int mt   = wave;
    const int ks   = blockIdx.x;        // K slice, 0..124
    const int m    = lane & 15;
    const int hi   = lane >> 4;

    const unsigned short* Arow = valsB + (long long)(mt * 16 + m) * NF;

    const v8f vz = {0, 0, 0, 0, 0, 0, 0, 0};
    v8f acc[8] = {vz, vz, vz, vz, vz, vz, vz, vz};

    const int off0 = hi ? 8 : 0;
    const int kc0 = ks * KC_PER;

    // prefetch chunk 0 of this slice (each thread copies 16 bytes -> 8 KB/block)
    async_copy16(ctxB + (long long)kc0 * 4096 + tid * 8, &Bbuf[0][tid * 8]);

#pragma unroll
    for (int i = 0; i < KC_PER; ++i) {
        const int kc  = kc0 + i;
        const int cur = i & 1;
        if (i + 1 < KC_PER) {
            async_copy16(ctxB + (long long)(kc + 1) * 4096 + tid * 8,
                         &Bbuf[cur ^ 1][tid * 8]);
            wait_async1();               // current chunk landed, prefetch in flight
        } else {
            wait_async0();
        }
        __syncthreads();                 // all waves' async portions visible

        const int kbase = kc * 32;
        union { uint4 u4[2]; v16bf v; } a;
        a.u4[0] = *reinterpret_cast<const uint4*>(Arow + kbase + off0);
        a.u4[1] = *reinterpret_cast<const uint4*>(Arow + kbase + off0 + 16);
#pragma unroll
        for (int t = 0; t < 8; ++t) {
            union { uint4 u4[2]; v16bf v; } b;
            const unsigned short* bp = &Bbuf[cur][(t * 32 + lane) * 16];
            b.u4[0] = *reinterpret_cast<const uint4*>(bp);
            b.u4[1] = *reinterpret_cast<const uint4*>(bp + 8);
            acc[t] = __builtin_amdgcn_wmma_f32_16x16x32_bf16(
                false, a.v, false, b.v, (short)0, acc[t], false, false);
        }
        __syncthreads();                 // reads done before buffer reuse
    }

#pragma unroll
    for (int t = 0; t < 8; ++t) {
#pragma unroll
        for (int r = 0; r < 8; ++r) {
            atomicAdd(&out[(mt * 16 + r + 8 * hi) * DIM + t * 16 + m], acc[t][r]);
        }
    }
}

// ---------------------------------------------------------------------------
extern "C" void kernel_launch(void* const* d_in, const int* in_sizes, int n_in,
                              void* d_out, int out_size, void* d_ws, size_t ws_size,
                              hipStream_t stream) {
    const float* values = (const float*)d_in[0];
    const float* feat   = (const float*)d_in[1];
    const float* hid    = (const float*)d_in[2];
    const float* Wk     = (const float*)d_in[3];
    const float* bias   = (const float*)d_in[4];
    const float* u      = (const float*)d_in[5];
    const float* corr   = (const float*)d_in[6];
    const int*   part   = (const int*)d_in[7];
    const int*   conn   = (const int*)d_in[8];
    float* out = (float*)d_out;

    char* ws = (char*)d_ws;
    size_t off = 0;
    auto take = [&](size_t bytes) -> char* {
        char* p = ws + off;
        off = (off + bytes + 255) & ~(size_t)255;
        return p;
    };
    float*          scores = (float*)take((size_t)NE * 4);
    float*          denom  = (float*)take((size_t)NF * 4);
    float*          ctx    = (float*)take((size_t)NF * DIM * 4);
    unsigned short* valsB  = (unsigned short*)take((size_t)BATCH * NF * 2);
    unsigned short* ctxB   = (unsigned short*)take((size_t)NF * DIM * 2);
    (void)ws_size; (void)in_sizes; (void)n_in; (void)out_size;

    zero_kernel<<<(NF + 255) / 256, 256, 0, stream>>>(denom, (long long)NF);
    zero_kernel<<<((long long)NF * DIM + 255) / 256, 256, 0, stream>>>(ctx, (long long)NF * DIM);
    zero_kernel<<<(BATCH * DIM + 255) / 256, 256, 0, stream>>>(out, (long long)BATCH * DIM);

    // 640000 connections / (16 per wave * 8 waves per block) = 5000 blocks
    attn_scores_kernel<<<NE / (16 * 8), 256, 0, stream>>>(
        feat, hid, Wk, bias, u, corr, part, conn, scores, denom);

    scatter_context_kernel<<<NE / SCH, 128, 0, stream>>>(
        scores, denom, part, conn, feat, hid, ctx);

    pack_values_kernel<<<(BATCH * (long long)NF / 4) / 256, 256, 0, stream>>>(values, valsB);
    pack_ctxB_kernel<<<((long long)NF * DIM) / 256, 256, 0, stream>>>(ctx, ctxB);

    final_gemm_kernel<<<KSPLIT, 512, 0, stream>>>(valsB, ctxB, out);
}
